// DenseBlock_33655363731619
// MI455X (gfx1250) — matrix operand order repaired and checked by reading, not compiled, not run on hardware
//
#include <hip/hip_runtime.h>
#include <math.h>

// ---------------------------------------------------------------------------
// CDNA5 (gfx1250) wave32 WMMA types
// ---------------------------------------------------------------------------
typedef __attribute__((ext_vector_type(16))) __bf16   v16bf;
typedef __attribute__((ext_vector_type(8)))  float    v8f;
typedef __attribute__((ext_vector_type(8)))  unsigned v8u;

#if defined(__HIP_DEVICE_COMPILE__)
#if __has_builtin(__builtin_amdgcn_global_load_async_to_lds_b128) && \
    __has_builtin(__builtin_amdgcn_s_wait_asynccnt)
#define HAS_ASYNC 1
#endif
#endif
#ifndef HAS_ASYNC
#define HAS_ASYNC 0
#endif

__device__ __forceinline__ unsigned short f2bf(float f) {
  unsigned u = __builtin_bit_cast(unsigned, f);
  u += 0x7FFFu + ((u >> 16) & 1u);           // round-to-nearest-even
  return (unsigned short)(u >> 16);
}
__device__ __forceinline__ v16bf bfrag(v8u u) { return __builtin_bit_cast(v16bf, u); }

// 16-byte global->LDS copy: async DMA on CDNA5 if the builtin exists.
__device__ __forceinline__ void copy16_g2l(const unsigned short* g, unsigned short* l) {
#if HAS_ASYNC
  typedef __attribute__((vector_size(16))) int vi4_t;
  typedef __attribute__((address_space(1))) vi4_t gvi4;
  typedef __attribute__((address_space(3))) vi4_t lvi4;
  __builtin_amdgcn_global_load_async_to_lds_b128(
      (gvi4*)(unsigned long long)g,
      (lvi4*)(unsigned)(unsigned long long)l, 0, 0);
#else
  *(uint4*)l = *(const uint4*)g;
#endif
}
__device__ __forceinline__ void wait_async_all() {
#if HAS_ASYNC
  __builtin_amdgcn_s_wait_asynccnt(0);
#endif
}

// ---------------------------------------------------------------------------
// Generic bf16 WMMA GEMM:  out = act(A(MxK) @ W(NxK)^T + bias) + res
// block = 256 threads = 8 waves, tile 128x128, BK = 32.
// Wave (wm in 0..1, wn in 0..3) owns a 64x32 region = 4x2 wmma tiles.
// Full interior tiles use a double-buffered async global->LDS pipeline;
// ragged tiles (N tails, odd K) use the bounds-checked synchronous path.
// ---------------------------------------------------------------------------
template<bool GELU, bool OUT_BF16>
__global__ __launch_bounds__(256) void gemm_wmma(
    const unsigned short* __restrict__ A, const unsigned short* __restrict__ W,
    const float* __restrict__ bias, const float* __restrict__ res,
    void* __restrict__ outp, int M, int N, int K)
{
  __shared__ unsigned short As[2 * 4096];   // 2 x (128 x 32) bf16
  __shared__ unsigned short Bs[2 * 4096];

  const int tid  = threadIdx.x;
  const int lane = tid & 31, wave = tid >> 5;
  const int wm = wave >> 2, wn = wave & 3;
  const int lm = lane & 15, lg = lane >> 4;
  const int m0 = blockIdx.y * 128, n0 = blockIdx.x * 128;

  v8f acc[4][2] = {};
  const int KT = (K + 31) >> 5;

  // Fragment gather + 8 WMMAs on one LDS buffer pair.
  auto compute = [&](const unsigned short* Asb, const unsigned short* Bsb) {
    v16bf af[4];
#pragma unroll
    for (int i = 0; i < 4; ++i) {
      const int mb = (wm << 6) + (i << 4) + lm;
      v8u u;
#pragma unroll
      for (int jj = 0; jj < 8; ++jj) {
        const int Kix = (jj < 4) ? ((lg << 3) + (jj << 1))
                                 : (16 + (lg << 3) + ((jj - 4) << 1));
        u[jj] = *(const unsigned*)(Asb + (mb << 5) + Kix);
      }
      af[i] = bfrag(u);
    }
    v16bf bf2[2];
#pragma unroll
    for (int t = 0; t < 2; ++t) {
      const int nb = (wn << 5) + (t << 4) + lm;
      v8u u;
#pragma unroll
      for (int jj = 0; jj < 8; ++jj) {
        const int Kix = (jj < 4) ? ((lg << 3) + (jj << 1))
                                 : (16 + (lg << 3) + ((jj - 4) << 1));
        u[jj] = *(const unsigned*)(Bsb + (nb << 5) + Kix);
      }
      bf2[t] = bfrag(u);
    }
#pragma unroll
    for (int i = 0; i < 4; ++i)
#pragma unroll
      for (int t = 0; t < 2; ++t)
        acc[i][t] = __builtin_amdgcn_wmma_f32_16x16x32_bf16(
            false, af[i], false, bf2[t], (short)0, acc[i][t], false, false);
  };

  const bool fast = (m0 + 128 <= M) && (n0 + 128 <= N) && ((K & 31) == 0);
  if (fast) {
    // ---- async double-buffered pipeline ----
    auto issue = [&](int kt, int buf) {
      const int k0 = kt << 5;
#pragma unroll
      for (int p = 0; p < 2; ++p) {
        const int j   = tid + (p << 8);      // 0..511 16B-chunks per tile
        const int row = j >> 2;
        const int c8  = (j & 3) << 3;
        copy16_g2l(A + (size_t)(m0 + row) * K + k0 + c8,
                   As + buf * 4096 + (row << 5) + c8);
        copy16_g2l(W + (size_t)(n0 + row) * K + k0 + c8,
                   Bs + buf * 4096 + (row << 5) + c8);
      }
    };
    issue(0, 0);
    wait_async_all();
    __syncthreads();
    for (int kt = 0; kt < KT; ++kt) {
      const int cur = kt & 1;
      if (kt + 1 < KT) issue(kt + 1, cur ^ 1);   // overlap with WMMAs below
      compute(As + cur * 4096, Bs + cur * 4096);
      if (kt + 1 < KT) { wait_async_all(); __syncthreads(); }
    }
  } else {
    // ---- bounds-checked synchronous path (buffer 0 only) ----
    for (int kt = 0; kt < KT; ++kt) {
      const int k0 = kt << 5;
      __syncthreads();
#pragma unroll
      for (int p = 0; p < 8; ++p) {
        const int j   = tid + p * 256;       // 0..2047 dwords per tile
        const int row = j >> 4;
        const int c2  = (j & 15) << 1;
        const int gk  = k0 + c2;
        const int gm  = m0 + row;
        unsigned va = 0u;
        if (gm < M && gk < K) va = *(const unsigned*)(A + (size_t)gm * K + gk);
        *(unsigned*)(As + (row << 5) + c2) = va;
        const int gn = n0 + row;
        unsigned vb = 0u;
        if (gn < N && gk < K) vb = *(const unsigned*)(W + (size_t)gn * K + gk);
        *(unsigned*)(Bs + (row << 5) + c2) = vb;
      }
      __syncthreads();
      compute(As, Bs);
    }
  }

  // Epilogue (C layout: VGPR r -> M = r + 8*g, N = lane%16)
#pragma unroll
  for (int i = 0; i < 4; ++i)
#pragma unroll
    for (int t = 0; t < 2; ++t)
#pragma unroll
      for (int r = 0; r < 8; ++r) {
        const int gm = m0 + (wm << 6) + (i << 4) + r + (lg << 3);
        const int gn = n0 + (wn << 5) + (t << 4) + lm;
        if (gm < M && gn < N) {
          float v = acc[i][t][r];
          if (bias) v += bias[gn];
          if (GELU) v = 0.5f * v * (1.0f + erff(v * 0.70710678118654752f));
          if (res)  v += res[(size_t)gm * N + gn];
          if (OUT_BF16) ((unsigned short*)outp)[(size_t)gm * N + gn] = f2bf(v);
          else          ((float*)outp)[(size_t)gm * N + gn]          = v;
        }
      }
}

// ---------------------------------------------------------------------------
// Stage-1 attention, WMMA (full score row block in registers).
// One wave handles (b, h, 16-query tile). 13 key tiles (208 >= 197).
// qkv: (25216, 2304) bf16, q at col h*64, k at 768+h*64, v at 1536+h*64.
// o  : (25216, 768) bf16.
// ---------------------------------------------------------------------------
__global__ __launch_bounds__(128) void attn_kernel(
    const unsigned short* __restrict__ qkv, unsigned short* __restrict__ o)
{
  __shared__ unsigned short P[4][16][224];   // per-wave probability tile
  const int tid  = threadIdx.x;
  const int lane = tid & 31, w = tid >> 5;
  const int task = blockIdx.x * 4 + w;       // grid sized exactly: B*H*13
  const int qt   = task % 13;
  const int bh   = task / 13;
  const int b    = bh / 12, h = bh % 12;
  const int lm = lane & 15, lg = lane >> 4;
  const int q0 = qt * 16;

  const unsigned short* qb = qkv + (size_t)b * 197 * 2304 + h * 64;
  const unsigned short* kb = qb + 768;
  const unsigned short* vb = qb + 1536;

  // ---- S = Q K^T -----------------------------------------------------------
  v8f S[13] = {};
#pragma unroll
  for (int ks = 0; ks < 2; ++ks) {           // head dim 64 = 2 x K32
    const int  qtok = q0 + lm;
    const bool qv   = qtok < 197;
    const unsigned short* qrow = qb + (size_t)(qv ? qtok : 0) * 2304 + ks * 32;
    v8u ua;
#pragma unroll
    for (int jj = 0; jj < 8; ++jj) {
      const int Kix = (jj < 4) ? ((lg << 3) + (jj << 1))
                               : (16 + (lg << 3) + ((jj - 4) << 1));
      ua[jj] = qv ? *(const unsigned*)(qrow + Kix) : 0u;
    }
    const v16bf af = bfrag(ua);
#pragma unroll
    for (int kt = 0; kt < 13; ++kt) {
      const int  ktok = kt * 16 + lm;
      const bool kv   = ktok < 197;
      const unsigned short* krow = kb + (size_t)(kv ? ktok : 0) * 2304 + ks * 32;
      v8u ub;
#pragma unroll
      for (int jj = 0; jj < 8; ++jj) {
        const int Kix = (jj < 4) ? ((lg << 3) + (jj << 1))
                                 : (16 + (lg << 3) + ((jj - 4) << 1));
        ub[jj] = kv ? *(const unsigned*)(krow + Kix) : 0u;
      }
      S[kt] = __builtin_amdgcn_wmma_f32_16x16x32_bf16(
          false, af, false, bfrag(ub), (short)0, S[kt], false, false);
    }
  }

  // ---- softmax (row-wise; each row lives in one 16-lane group) -------------
  const float scale = 0.125f;                // 64^-0.5
  float mx[8], sm[8], rinv[8];
#pragma unroll
  for (int r = 0; r < 8; ++r) mx[r] = -3.0e38f;
#pragma unroll
  for (int kt = 0; kt < 13; ++kt) {
    const bool cv = (kt * 16 + lm) < 197;    // mask padded key columns
#pragma unroll
    for (int r = 0; r < 8; ++r) {
      float s = cv ? S[kt][r] * scale : -3.0e38f;
      S[kt][r] = s;
      mx[r] = fmaxf(mx[r], s);
    }
  }
#pragma unroll
  for (int r = 0; r < 8; ++r)
#pragma unroll
    for (int m = 1; m < 16; m <<= 1) mx[r] = fmaxf(mx[r], __shfl_xor(mx[r], m, 32));
#pragma unroll
  for (int r = 0; r < 8; ++r) sm[r] = 0.0f;
#pragma unroll
  for (int kt = 0; kt < 13; ++kt)
#pragma unroll
    for (int r = 0; r < 8; ++r) {
      float p = __expf(S[kt][r] - mx[r]);
      S[kt][r] = p;
      sm[r] += p;
    }
#pragma unroll
  for (int r = 0; r < 8; ++r) {
#pragma unroll
    for (int m = 1; m < 16; m <<= 1) sm[r] += __shfl_xor(sm[r], m, 32);
    rinv[r] = 1.0f / sm[r];
  }

  // ---- stage normalized P in LDS (bf16), re-gather as A fragments ----------
#pragma unroll
  for (int kt = 0; kt < 13; ++kt) {
    const int col = kt * 16 + lm;
#pragma unroll
    for (int r = 0; r < 8; ++r)
      P[w][r + (lg << 3)][col] = f2bf(S[kt][r] * rinv[r]);
  }
  { // zero pad columns 208..223
    const int col = 208 + lm;
#pragma unroll
    for (int r = 0; r < 8; ++r) P[w][r + (lg << 3)][col] = 0;
  }
  __syncthreads();

  // ---- O = P V -------------------------------------------------------------
  v8f O[4] = {};
#pragma unroll
  for (int k2 = 0; k2 < 7; ++k2) {           // 224 keys = 7 x K32
    v8u ua;
#pragma unroll
    for (int jj = 0; jj < 8; ++jj) {
      const int Kix = (jj < 4) ? ((lg << 3) + (jj << 1))
                               : (16 + (lg << 3) + ((jj - 4) << 1));
      ua[jj] = *(const unsigned*)(&P[w][lm][k2 * 32 + Kix]);
    }
    const v16bf af = bfrag(ua);
#pragma unroll
    for (int nt = 0; nt < 4; ++nt) {
      v8u ub;
#pragma unroll
      for (int jj = 0; jj < 8; ++jj) {
        const int kk = k2 * 32 + ((jj < 4) ? ((lg << 3) + (jj << 1))
                                           : (16 + (lg << 3) + ((jj - 4) << 1)));
        const int d  = nt * 16 + lm;
        unsigned lo = (kk     < 197) ? (unsigned)vb[(size_t)kk * 2304 + d]       : 0u;
        unsigned hi = (kk + 1 < 197) ? (unsigned)vb[(size_t)(kk + 1) * 2304 + d] : 0u;
        ub[jj] = lo | (hi << 16);
      }
      O[nt] = __builtin_amdgcn_wmma_f32_16x16x32_bf16(
          false, af, false, bfrag(ub), (short)0, O[nt], false, false);
    }
  }
#pragma unroll
  for (int nt = 0; nt < 4; ++nt)
#pragma unroll
    for (int r = 0; r < 8; ++r) {
      const int tok = q0 + r + (lg << 3);
      if (tok < 197)
        o[(size_t)(b * 197 + tok) * 768 + h * 64 + nt * 16 + lm] = f2bf(O[nt][r]);
    }
}

// ---------------------------------------------------------------------------
// LayerNorm over C=768, f32 in -> bf16 out. One row per 256-thread block.
// ---------------------------------------------------------------------------
__global__ __launch_bounds__(256) void ln768_kernel(
    const float* __restrict__ x, const float* __restrict__ g,
    const float* __restrict__ b, unsigned short* __restrict__ out)
{
  __shared__ float red[256];
  const int row = blockIdx.x, tid = threadIdx.x;
  const float* xr = x + (size_t)row * 768;
  const float v0 = xr[tid], v1 = xr[tid + 256], v2 = xr[tid + 512];
  red[tid] = v0 + v1 + v2;
  __syncthreads();
  for (int s = 128; s > 0; s >>= 1) { if (tid < s) red[tid] += red[tid + s]; __syncthreads(); }
  const float mean = red[0] * (1.0f / 768.0f);
  __syncthreads();
  const float d0 = v0 - mean, d1 = v1 - mean, d2 = v2 - mean;
  red[tid] = d0 * d0 + d1 * d1 + d2 * d2;
  __syncthreads();
  for (int s = 128; s > 0; s >>= 1) { if (tid < s) red[tid] += red[tid + s]; __syncthreads(); }
  const float rs = rsqrtf(red[0] * (1.0f / 768.0f) + 1e-5f);
  unsigned short* orow = out + (size_t)row * 768;
  orow[tid]       = f2bf(d0 * rs * g[tid]       + b[tid]);
  orow[tid + 256] = f2bf(d1 * rs * g[tid + 256] + b[tid + 256]);
  orow[tid + 512] = f2bf(d2 * rs * g[tid + 512] + b[tid + 512]);
}

// LayerNorm over C=32 -> bf16 row-major out. block (32,8), shuffle reduce.
__global__ __launch_bounds__(256) void ln32_kernel(
    const float* __restrict__ x, const float* __restrict__ g,
    const float* __restrict__ b, unsigned short* __restrict__ out, int M)
{
  const int row = blockIdx.x * 8 + threadIdx.y;
  const int c = threadIdx.x;
  if (row >= M) return;
  const float v = x[(size_t)row * 32 + c];
  float s = v;
#pragma unroll
  for (int m = 16; m; m >>= 1) s += __shfl_xor(s, m, 32);
  const float mean = s * (1.0f / 32.0f);
  const float d = v - mean;
  float q = d * d;
#pragma unroll
  for (int m = 16; m; m >>= 1) q += __shfl_xor(q, m, 32);
  const float rs = rsqrtf(q * (1.0f / 32.0f) + 1e-5f);
  out[(size_t)row * 32 + c] = f2bf(d * rs * g[c] + b[c]);
}

// LN3: C=32; writes cls (token 0, f32) and transposed t (B,32,196) bf16.
__global__ __launch_bounds__(256) void ln3_kernel(
    const float* __restrict__ x, const float* __restrict__ g,
    const float* __restrict__ b, float* __restrict__ cls,
    unsigned short* __restrict__ tbf, int M)
{
  const int row = blockIdx.x * 8 + threadIdx.y;
  const int c = threadIdx.x;
  if (row >= M) return;
  const float v = x[(size_t)row * 32 + c];
  float s = v;
#pragma unroll
  for (int m = 16; m; m >>= 1) s += __shfl_xor(s, m, 32);
  const float mean = s * (1.0f / 32.0f);
  const float d = v - mean;
  float q = d * d;
#pragma unroll
  for (int m = 16; m; m >>= 1) q += __shfl_xor(q, m, 32);
  const float rs = rsqrtf(q * (1.0f / 32.0f) + 1e-5f);
  const float val = d * rs * g[c] + b[c];
  const int bb = row / 197, t = row % 197;
  if (t == 0) cls[bb * 32 + c] = val;
  else        tbf[((size_t)bb * 32 + c) * 196 + (t - 1)] = f2bf(val);
}

// ---------------------------------------------------------------------------
// Channel attention (tiny: 32 tokens, dim 14, 14 heads) in VALU.
// One 32-thread block per (b, h); thread = query token.
// cqkv: (B*32, 588) f32, col = s*196 + h*14 + d.  co: (B*32, 196) bf16.
// ---------------------------------------------------------------------------
__global__ __launch_bounds__(32) void cattn_kernel(
    const float* __restrict__ cqkv, unsigned short* __restrict__ co)
{
  __shared__ float Ks[32][14], Vs[32][14];
  const int bh = blockIdx.x;
  const int b = bh / 14, h = bh % 14;
  const int t = threadIdx.x;
  const float* base = cqkv + (size_t)b * 32 * 588;
  float qv[14];
#pragma unroll
  for (int d = 0; d < 14; ++d) {
    qv[d]    = base[(size_t)t * 588 +       h * 14 + d];
    Ks[t][d] = base[(size_t)t * 588 + 196 + h * 14 + d];
    Vs[t][d] = base[(size_t)t * 588 + 392 + h * 14 + d];
  }
  __syncthreads();
  const float scale = 0.2672612419124244f;   // 14^-0.5
  float sc[32];
  float mx = -3.0e38f;
#pragma unroll
  for (int k = 0; k < 32; ++k) {
    float s = 0.0f;
#pragma unroll
    for (int d = 0; d < 14; ++d) s += qv[d] * Ks[k][d];
    s *= scale;
    sc[k] = s;
    mx = fmaxf(mx, s);
  }
  float sum = 0.0f;
#pragma unroll
  for (int k = 0; k < 32; ++k) { sc[k] = __expf(sc[k] - mx); sum += sc[k]; }
  const float inv = 1.0f / sum;
#pragma unroll
  for (int d = 0; d < 14; ++d) {
    float ov = 0.0f;
#pragma unroll
    for (int k = 0; k < 32; ++k) ov += sc[k] * Vs[k][d];
    co[((size_t)b * 32 + t) * 196 + h * 14 + d] = f2bf(ov * inv);
  }
}

// x3 = x2 + concat(cls, cop^T) over (B,197,32)
__global__ __launch_bounds__(256) void assemble_kernel(
    const float* __restrict__ x2, const float* __restrict__ cls,
    const float* __restrict__ cop, float* __restrict__ x3)
{
  const int i = blockIdx.x * 256 + threadIdx.x;
  if (i >= 128 * 197 * 32) return;
  const int c = i & 31;
  const int rem = i >> 5;
  const int t = rem % 197, bb = rem / 197;
  const float add = (t == 0) ? cls[bb * 32 + c]
                             : cop[((size_t)bb * 32 + c) * 196 + (t - 1)];
  x3[i] = x2[i] + add;
}

__global__ __launch_bounds__(256) void cast_bf16_kernel(
    const float* __restrict__ in, unsigned short* __restrict__ out, int n)
{
  const int i = blockIdx.x * 256 + threadIdx.x;
  if (i < n) out[i] = f2bf(in[i]);
}

// ---------------------------------------------------------------------------
// Host launcher
// ---------------------------------------------------------------------------
extern "C" void kernel_launch(void* const* d_in, const int* in_sizes, int n_in,
                              void* d_out, int out_size, void* d_ws, size_t ws_size,
                              hipStream_t stream) {
  (void)in_sizes; (void)n_in; (void)out_size; (void)ws_size;

  const float* x      = (const float*)d_in[0];
  const float* g1     = (const float*)d_in[1];
  const float* b1     = (const float*)d_in[2];
  const float* w_qkv  = (const float*)d_in[3];
  const float* w_proj = (const float*)d_in[4];
  const float* b_proj = (const float*)d_in[5];
  const float* g2     = (const float*)d_in[6];
  const float* b2     = (const float*)d_in[7];
  const float* w_fc1  = (const float*)d_in[8];
  const float* b_fc1  = (const float*)d_in[9];
  const float* w_fc2  = (const float*)d_in[10];
  const float* b_fc2  = (const float*)d_in[11];
  const float* g3     = (const float*)d_in[12];
  const float* b3     = (const float*)d_in[13];
  const float* w_cqkv = (const float*)d_in[14];
  const float* w_cproj= (const float*)d_in[15];
  const float* b_cproj= (const float*)d_in[16];
  const float* g4     = (const float*)d_in[17];
  const float* b4     = (const float*)d_in[18];
  const float* w_cfc1 = (const float*)d_in[19];
  const float* b_cfc1 = (const float*)d_in[20];
  const float* w_cfc2 = (const float*)d_in[21];
  const float* b_cfc2 = (const float*)d_in[22];

  const int M1 = 128 * 197;                  // 25216 tokens

  // --- workspace carve (deterministic) ---
  char* wsp = (char*)d_ws;
  size_t off = 0;
  auto take = [&](size_t bytes) -> void* {
    void* p = wsp + off;
    off = (off + bytes + 255) & ~(size_t)255;
    return p;
  };
  // region A: qkv (M1 x 2304) then reused for h1 (M1 x 3072)
  unsigned short* regA = (unsigned short*)take((size_t)M1 * 3072 * 2);
  unsigned short* qkv  = regA;
  unsigned short* h1   = regA;
  // region B: ln1 -> o -> ln2 (M1 x 768 bf16)
  unsigned short* regB = (unsigned short*)take((size_t)M1 * 768 * 2);
  unsigned short* ln1b = regB, *obf = regB, *ln2b = regB;
  float*          x1   = (float*)take((size_t)M1 * 768 * 4);
  float*          x2   = (float*)take((size_t)M1 * 32 * 4);
  float*          cls  = (float*)take((size_t)128 * 32 * 4);
  unsigned short* tbf  = (unsigned short*)take((size_t)128 * 32 * 196 * 2);
  float*          cqkv = (float*)take((size_t)128 * 32 * 588 * 4);
  unsigned short* cobf = (unsigned short*)take((size_t)128 * 32 * 196 * 2);
  float*          cop  = (float*)take((size_t)128 * 32 * 196 * 4);
  float*          x3   = (float*)take((size_t)M1 * 32 * 4);
  unsigned short* ln4b = (unsigned short*)take((size_t)M1 * 32 * 2);
  unsigned short* hc   = (unsigned short*)take((size_t)M1 * 128 * 2);
  unsigned short* wqkv_b  = (unsigned short*)take((size_t)2304 * 768 * 2);
  unsigned short* wproj_b = (unsigned short*)take((size_t)768 * 768 * 2);
  unsigned short* wfc1_b  = (unsigned short*)take((size_t)3072 * 768 * 2);
  unsigned short* wfc2_b  = (unsigned short*)take((size_t)32 * 3072 * 2);
  unsigned short* wcqkv_b = (unsigned short*)take((size_t)588 * 196 * 2);
  unsigned short* wcproj_b= (unsigned short*)take((size_t)196 * 196 * 2);
  unsigned short* wcfc1_b = (unsigned short*)take((size_t)128 * 32 * 2);
  unsigned short* wcfc2_b = (unsigned short*)take((size_t)32 * 128 * 2);

  auto cast = [&](const float* s, unsigned short* dst, int n) {
    cast_bf16_kernel<<<(n + 255) / 256, 256, 0, stream>>>(s, dst, n);
  };
  cast(w_qkv,  wqkv_b,  2304 * 768);
  cast(w_proj, wproj_b, 768 * 768);
  cast(w_fc1,  wfc1_b,  3072 * 768);
  cast(w_fc2,  wfc2_b,  32 * 3072);
  cast(w_cqkv, wcqkv_b, 588 * 196);
  cast(w_cproj,wcproj_b,196 * 196);
  cast(w_cfc1, wcfc1_b, 128 * 32);
  cast(w_cfc2, wcfc2_b, 32 * 128);

  // Stage 1: LN1 -> QKV -> attention -> proj (+x)
  ln768_kernel<<<M1, 256, 0, stream>>>(x, g1, b1, ln1b);
  gemm_wmma<false, true ><<<dim3(18, 197), 256, 0, stream>>>(ln1b, wqkv_b, nullptr, nullptr, qkv, M1, 2304, 768);
  attn_kernel<<<4992, 128, 0, stream>>>(qkv, obf);                     // 128*12*13/4 waves
  gemm_wmma<false, false><<<dim3(6, 197), 256, 0, stream>>>(obf, wproj_b, b_proj, x, x1, M1, 768, 768);

  // Stage 2: LN2 -> fc1(gelu) -> fc2  (x replaced, no residual)
  ln768_kernel<<<M1, 256, 0, stream>>>(x1, g2, b2, ln2b);
  gemm_wmma<true,  true ><<<dim3(24, 197), 256, 0, stream>>>(ln2b, wfc1_b, b_fc1, nullptr, h1, M1, 3072, 768);
  gemm_wmma<false, false><<<dim3(1, 197), 256, 0, stream>>>(h1, wfc2_b, b_fc2, nullptr, x2, M1, 32, 3072);

  // Stage 3: LN3 -> channel attention -> cproj -> concat residual
  ln3_kernel<<<M1 / 8, dim3(32, 8), 0, stream>>>(x2, g3, b3, cls, tbf, M1);
  gemm_wmma<false, false><<<dim3(5, 32), 256, 0, stream>>>(tbf, wcqkv_b, nullptr, nullptr, cqkv, 128 * 32, 588, 196);
  cattn_kernel<<<128 * 14, 32, 0, stream>>>(cqkv, cobf);
  gemm_wmma<false, false><<<dim3(2, 32), 256, 0, stream>>>(cobf, wcproj_b, b_cproj, nullptr, cop, 128 * 32, 196, 196);
  assemble_kernel<<<(M1 * 32 + 255) / 256, 256, 0, stream>>>(x2, cls, cop, x3);

  // Stage 4: LN4 -> cfc1(gelu) -> cfc2 (+x3) -> d_out
  ln32_kernel<<<M1 / 8, dim3(32, 8), 0, stream>>>(x3, g4, b4, ln4b, M1);
  gemm_wmma<true,  true ><<<dim3(1, 197), 256, 0, stream>>>(ln4b, wcfc1_b, b_cfc1, nullptr, hc, M1, 128, 32);
  gemm_wmma<false, false><<<dim3(1, 197), 256, 0, stream>>>(hc, wcfc2_b, b_cfc2, x3, (float*)d_out, M1, 32, 128);
}
